// Attention_65816078844162
// MI455X (gfx1250) — matrix operand order repaired and checked by reading, not compiled, not run on hardware
//
#include <hip/hip_runtime.h>
#include <hip/hip_bf16.h>

// ---------------------------------------------------------------------------
// Problem constants (B,S,D,H from the reference)
// ---------------------------------------------------------------------------
static constexpr int Bb = 8;
static constexpr int Ss = 2048;
static constexpr int Dd = 1024;
static constexpr int Hh = 1024;

typedef __attribute__((ext_vector_type(16))) __bf16 v16bf;
typedef __attribute__((ext_vector_type(8)))  __bf16 v8bf;
typedef __attribute__((ext_vector_type(8)))  float  v8f;
typedef __attribute__((ext_vector_type(4)))  unsigned int v4u;
typedef __attribute__((ext_vector_type(8)))  unsigned int v8u;

__device__ __forceinline__ v8f wmma_bf16(v16bf a, v16bf b, v8f c) {
    // v_wmma_f32_16x16x32_bf16  (neg_a, A, neg_b, B, c_mod, C, reuse_a, reuse_b)
    return __builtin_amdgcn_wmma_f32_16x16x32_bf16(false, a, false, b, (short)0, c,
                                                   false, false);
}

// A fragment: 16(M) x 32(K) bf16, row-major with row stride `ld` (global or LDS).
// Lane layout: M=lane&15, K = hi*16 .. hi*16+15; hi = lane>>4.
// One contiguous 32-byte load per lane.
__device__ __forceinline__ v16bf load_frag_a(const __bf16* p, int ld, int lane) {
    const int row = lane & 15;
    const int hi  = lane >> 4;
    return *(const v16bf*)(p + (size_t)row * ld + hi * 16);
}

// B fragment: 32(K) x 16(N) where memory holds the N dimension as rows of
// length-K (i.e. B[k][n] = row n, element k : weight / key layout).
// Lane layout: lane holds column N=lane&15; lanes 0-15 -> K=0..15,
// lanes 16-31 -> K=16..31. One contiguous 32-byte load per lane.
__device__ __forceinline__ v16bf load_frag_b_rowk(const __bf16* __restrict__ p, int ld, int lane) {
    const int col = lane & 15;
    const int hi  = lane >> 4;
    return *(const v16bf*)(p + (size_t)col * ld + hi * 16);
}

// ---------------------------------------------------------------------------
// Kernel 1: fp32 -> bf16 conversion (grid-stride)
// ---------------------------------------------------------------------------
__global__ void __launch_bounds__(256)
cvt_f32_bf16(const float* __restrict__ in, __bf16* __restrict__ out, int n) {
    int i = blockIdx.x * blockDim.x + threadIdx.x;
    const int stride = gridDim.x * blockDim.x;
    for (; i < n; i += stride) out[i] = (__bf16)in[i];
}

// ---------------------------------------------------------------------------
// Kernel 2: C[m,n] = sum_k A[m,k] * W[n,k] + bias[n]   (nn.Linear, y = x W^T + b)
// A: M x K bf16 row-major, W: N x K bf16 row-major, C: M x N bf16.
// Workgroup: 8 waves (4 in M x 2 in N); wave tile 32x64 -> block tile 128x128.
// ---------------------------------------------------------------------------
__global__ void __launch_bounds__(256)
gemm_xwT_bias(const __bf16* __restrict__ A, const __bf16* __restrict__ W,
              const float* __restrict__ bias, __bf16* __restrict__ C,
              int M, int N, int K) {
    const int lane  = threadIdx.x & 31;
    const int wave  = threadIdx.x >> 5;
    const int waveM = wave & 3;
    const int waveN = wave >> 2;
    const int m0 = blockIdx.y * 128 + waveM * 32;
    const int n0 = blockIdx.x * 128 + waveN * 64;
    const int hi = lane >> 4, lo = lane & 15;

    v8f acc[2][4];
#pragma unroll
    for (int i = 0; i < 2; ++i)
#pragma unroll
        for (int j = 0; j < 4; ++j)
#pragma unroll
            for (int r = 0; r < 8; ++r) acc[i][j][r] = 0.0f;

    for (int k0 = 0; k0 < K; k0 += 32) {
        // Unconditional speculative prefetch of the next K-slab
        // (global_prefetch_b8; silently dropped if the address is invalid).
        __builtin_prefetch(A + (size_t)(m0 + lo) * K + k0 + 32, 0, 1);
        __builtin_prefetch(W + (size_t)(n0 + lo) * K + k0 + 32, 0, 1);

        v16bf a[2], b[4];
#pragma unroll
        for (int i = 0; i < 2; ++i)
            a[i] = load_frag_a(A + (size_t)(m0 + i * 16) * K + k0, K, lane);
#pragma unroll
        for (int j = 0; j < 4; ++j)
            b[j] = load_frag_b_rowk(W + (size_t)(n0 + j * 16) * K + k0, K, lane);
#pragma unroll
        for (int i = 0; i < 2; ++i)
#pragma unroll
            for (int j = 0; j < 4; ++j)
                acc[i][j] = wmma_bf16(a[i], b[j], acc[i][j]);
    }

#pragma unroll
    for (int j = 0; j < 4; ++j) {
        const float bj = bias[n0 + j * 16 + lo];
#pragma unroll
        for (int i = 0; i < 2; ++i) {
#pragma unroll
            for (int r = 0; r < 8; ++r) {
                const int mrow = m0 + i * 16 + r + 8 * hi;
                C[(size_t)mrow * N + n0 + j * 16 + lo] = (__bf16)(acc[i][j][r] + bj);
            }
        }
    }
}

// ---------------------------------------------------------------------------
// Kernel 3: causal softmax. One wave owns a 16-query tile.
// Q tile (16x1024 bf16, 32KB) is async-staged into LDS once per wave with
// GLOBAL_LOAD_ASYNC_TO_LDS_B128 (wave-private -> only s_wait_asynccnt needed),
// so the ~qt key-tile iterations read A-fragments from LDS instead of
// re-streaming Q from L2. Raw fp32 scores cached in LDS (128KB/wave).
// LDS: 2*(128KB + 32KB) = 320KB (the per-workgroup cap).
// Row stats via half-wave __shfl_xor reductions (wave32). P written as bf16,
// zero-padded to a 32-key boundary so the PV GEMM can use K=32 WMMA steps.
// ---------------------------------------------------------------------------
__global__ void __launch_bounds__(64)
attn_softmax(const __bf16* __restrict__ Q, const __bf16* __restrict__ Kmat,
             __bf16* __restrict__ P) {
    extern __shared__ char smem_raw[];  // [0,256KB): scores f32; [256KB,320KB): Q bf16
    const int lane = threadIdx.x & 31;
    const int wave = threadIdx.x >> 5;
    const int tile = blockIdx.x * 2 + wave;     // global 16-row query tile
    const int b  = tile >> 7;                   // / (Ss/16)
    const int qt = tile & 127;                  // % (Ss/16)
    const int q0 = qt * 16;
    const int hi = lane >> 4, lo = lane & 15;

    float*  srow  = (float*)smem_raw + (size_t)wave * 16 * Ss;
    __bf16* qtile = (__bf16*)(smem_raw + 2u * 16u * Ss * 4u) + (size_t)wave * 16 * Hh;
    const unsigned int q_lds0 = (unsigned int)(size_t)(void*)qtile;

    const __bf16* Qb = Q    + (size_t)b * Ss * Hh;
    const __bf16* Kb = Kmat + (size_t)b * Ss * Hh;

    // Async-stage this wave's 16x1024 Q tile into LDS (2048 16B chunks).
#pragma unroll 4
    for (int i = 0; i < 64; ++i) {
        const int idx = lane + i * 32;     // chunk id 0..2047
        const int row = idx >> 7;          // 1024/8 = 128 chunks per row
        const int c8  = idx & 127;
        const __bf16* gp = Qb + (size_t)(q0 + row) * Hh + c8 * 8;
        const unsigned int lds = q_lds0 + (unsigned int)(row * Hh + c8 * 8) * 2u;
        asm volatile("global_load_async_to_lds_b128 %0, %1, off"
                     :: "v"(lds), "v"((unsigned long long)(size_t)gp)
                     : "memory");
    }
    asm volatile("s_wait_asynccnt 0" ::: "memory");  // wave-private: no barrier

    float mrow[8];
#pragma unroll
    for (int r = 0; r < 8; ++r) mrow[r] = -1e30f;

    const float scale = 1.0f / 32.0f;  // 1/sqrt(H), H=1024
    const int nkt = qt + 1;            // causal: key tiles 0..qt

    for (int kt = 0; kt < nkt; ++kt) {
        const int k0 = kt * 16;
        v8f acc;
#pragma unroll
        for (int r = 0; r < 8; ++r) acc[r] = 0.0f;

        for (int h0 = 0; h0 < Hh; h0 += 32) {
            v16bf a  = load_frag_a(qtile + h0, Hh, lane);  // from LDS
            v16bf bb = load_frag_b_rowk(Kb + (size_t)k0 * Hh + h0, Hh, lane);
            acc = wmma_bf16(a, bb, acc);
        }

#pragma unroll
        for (int r = 0; r < 8; ++r) {
            float s = acc[r] * scale;
            const int kg = k0 + lo;
            const int qg = q0 + r + 8 * hi;
            if (kg > qg) s = -1e30f;               // causal mask
            srow[(r + 8 * hi) * Ss + kg] = s;
            float v = s;                           // running row max (16 lanes)
#pragma unroll
            for (int off = 8; off; off >>= 1) v = fmaxf(v, __shfl_xor(v, off, 32));
            mrow[r] = fmaxf(mrow[r], v);
        }
    }

    const int kmax = q0 + 16;  // keys materialized for this tile

    float invl[8];
#pragma unroll
    for (int r = 0; r < 8; ++r) {
        float sum = 0.0f;
        for (int k = lo; k < kmax; k += 16)
            sum += __expf(srow[(r + 8 * hi) * Ss + k] - mrow[r]);
#pragma unroll
        for (int off = 8; off; off >>= 1) sum += __shfl_xor(sum, off, 32);
        invl[r] = 1.0f / sum;
    }

    __bf16* Pb = P + (size_t)b * Ss * Ss;
#pragma unroll
    for (int r = 0; r < 8; ++r) {
        const int mq = r + 8 * hi;
        for (int k = lo; k < kmax; k += 16) {
            const float p = __expf(srow[mq * Ss + k] - mrow[r]) * invl[r];
            Pb[(size_t)(q0 + mq) * Ss + k] = (__bf16)p;
        }
        if (kmax & 31) {  // pad the next 16-key tile with zeros (32-key alignment)
            Pb[(size_t)(q0 + mq) * Ss + kmax + lo] = (__bf16)0.0f;
        }
    }
}

// ---------------------------------------------------------------------------
// Kernel 4: O[q,h] = sum_k P[q,k] * V[k,h]  (fp32 output = final result).
// 4 waves per WG, each wave: 16 queries x 64 output columns (4 WMMA C tiles).
//
// V tile path (CDNA5 TDM): each K-step, wave 0 issues ONE tensor_load_to_lds
// with a 2D D# (tile 256(x) x 32(y) of the 2048x1024 bf16 V tensor, dim0
// stride = H) that DMAs the whole 16KB tile into LDS, drained with
// s_wait_tensorcnt and published by a barrier. The tile is then transposed in
// LDS to [h][k] (64B rows) so each WMMA B-fragment is a single aligned
// 32-byte LDS read per lane. Causal bound: k < round_up(q0+16,32) — P was
// zero-padded there.
// ---------------------------------------------------------------------------
__global__ void __launch_bounds__(128)
gemm_pv(const __bf16* __restrict__ P, const __bf16* __restrict__ V,
        float* __restrict__ O) {
    __shared__ __bf16 rowbuf[32 * 256];    // V tile, row-major [k][h]   (16 KB)
    __shared__ __bf16 transbuf[256 * 32];  // V tile, transposed [h][k]  (16 KB)

    const int tid  = threadIdx.x;
    const int lane = tid & 31;
    const int wave = tid >> 5;
    const int b  = blockIdx.z;
    const int qt = blockIdx.y;
    const int q0 = qt * 16;
    const int h0g = blockIdx.x * 256;   // workgroup's H range
    const int hw  = wave * 64;          // wave's H offset inside the WG range
    const int hi = lane >> 4, lo = lane & 15;

    const __bf16* Pb = P + (size_t)b * Ss * Ss + (size_t)q0 * Ss;
    const __bf16* Vb = V + (size_t)b * Ss * Hh;

    // Raw 32-bit LDS byte address of rowbuf (flat LDS aperture keeps the LDS
    // offset in addr[31:0], ISA 10.2).
    const unsigned int row_lds0 = (unsigned int)(size_t)(void*)rowbuf;

    v8f acc[4];
#pragma unroll
    for (int j = 0; j < 4; ++j)
#pragma unroll
        for (int r = 0; r < 8; ++r) acc[j][r] = 0.0f;

    const int kend = ((q0 + 16) + 31) & ~31;
    for (int k0 = 0; k0 < kend; k0 += 32) {
        // 1) TDM: one DMA for the whole 32x256 V tile (wave 0 issues).
        if (wave == 0) {
            const unsigned long long gaddr =
                (unsigned long long)(size_t)(Vb + (size_t)k0 * Hh + h0g);
            v4u g0;
            g0[0] = 1u;                                   // count=1 (valid), user D#
            g0[1] = row_lds0;                             // lds_addr
            g0[2] = (unsigned int)(gaddr & 0xFFFFFFFFu);  // global_addr[31:0]
            g0[3] = (unsigned int)((gaddr >> 32) & 0x01FFFFFFu) | (2u << 30); // addr[56:32] | type=2
            v8u g1;
            g1[0] = 1u << 16;                             // data_size=1 (2 bytes)
            g1[1] = ((unsigned int)Hh & 0xFFFFu) << 16;   // tensor_dim0[15:0]=1024
            g1[2] = ((unsigned int)Hh >> 16) |            // tensor_dim0[31:16]
                    (((unsigned int)Ss & 0xFFFFu) << 16); // tensor_dim1[15:0]=2048
            g1[3] = ((unsigned int)Ss >> 16) |            // tensor_dim1[31:16]
                    (256u << 16);                         // tile_dim0=256
            g1[4] = 32u;                                  // tile_dim1=32, tile_dim2=0
            g1[5] = (unsigned int)Hh;                     // tensor_dim0_stride=1024
            g1[6] = 0u;                                   // stride hi / dim1_stride lo
            g1[7] = 0u;                                   // dim1_stride hi
            asm volatile("tensor_load_to_lds %0, %1" :: "s"(g0), "s"(g1) : "memory");
            __builtin_amdgcn_s_wait_tensorcnt((unsigned short)0);
        }
        __syncthreads();

        // 2) transpose in LDS: rowbuf[k][h] -> transbuf[h][k] (rows of 64B)
#pragma unroll
        for (int i = 0; i < 8; ++i) {
            const int idx  = tid + i * 128;
            const int row  = idx >> 5;
            const int col8 = idx & 31;
            v8bf v = *(const v8bf*)(rowbuf + row * 256 + col8 * 8);
#pragma unroll
            for (int e = 0; e < 8; ++e)
                transbuf[(col8 * 8 + e) * 32 + row] = v[e];
        }
        __syncthreads();

        // 3) WMMA: A from global P (contiguous per-lane 16B chunks),
        //    B as one aligned 32B LDS read per lane.
        const v16bf a = load_frag_a(Pb + k0, Ss, lane);
#pragma unroll
        for (int j = 0; j < 4; ++j) {
            const v16bf bb = *(const v16bf*)(transbuf + (hw + j * 16 + lo) * 32 + hi * 16);
            acc[j] = wmma_bf16(a, bb, acc[j]);
        }
        __syncthreads();  // protect rowbuf/transbuf before next stage
    }

    float* Ob = O + (size_t)b * Ss * Hh + (size_t)q0 * Hh;
#pragma unroll
    for (int j = 0; j < 4; ++j)
#pragma unroll
        for (int r = 0; r < 8; ++r)
            Ob[(size_t)(r + 8 * hi) * Hh + h0g + hw + j * 16 + lo] = acc[j][r];
}

// ---------------------------------------------------------------------------
// Host launcher.
// Workspace layout (all 256B aligned):
//   x_bf   : B*S*D bf16   (32 MiB)
//   Wq/Wk/Wv bf16         (3 x 2 MiB)
//   q/k/v  : B*S*H bf16   (3 x 32 MiB)
//   P      : B*S*S bf16   (64 MiB)
// Total ~198 MiB.
// ---------------------------------------------------------------------------
extern "C" void kernel_launch(void* const* d_in, const int* in_sizes, int n_in,
                              void* d_out, int out_size, void* d_ws, size_t ws_size,
                              hipStream_t stream) {
    (void)in_sizes; (void)n_in; (void)out_size; (void)ws_size;

    const float* x  = (const float*)d_in[0];
    const float* Wq = (const float*)d_in[1];
    const float* bq = (const float*)d_in[2];
    const float* Wk = (const float*)d_in[3];
    const float* bk = (const float*)d_in[4];
    const float* Wv = (const float*)d_in[5];
    const float* bv = (const float*)d_in[6];
    float* out = (float*)d_out;

    char* ws = (char*)d_ws;
    size_t off = 0;
    auto alloc = [&](size_t bytes) -> void* {
        void* p = ws + off;
        off += (bytes + 255) & ~(size_t)255;
        return p;
    };

    const int M = Bb * Ss;  // 16384
    __bf16* x_bf  = (__bf16*)alloc((size_t)M * Dd * 2);
    __bf16* wq_bf = (__bf16*)alloc((size_t)Hh * Dd * 2);
    __bf16* wk_bf = (__bf16*)alloc((size_t)Hh * Dd * 2);
    __bf16* wv_bf = (__bf16*)alloc((size_t)Hh * Dd * 2);
    __bf16* q_bf  = (__bf16*)alloc((size_t)M * Hh * 2);
    __bf16* k_bf  = (__bf16*)alloc((size_t)M * Hh * 2);
    __bf16* v_bf  = (__bf16*)alloc((size_t)M * Hh * 2);
    __bf16* p_bf  = (__bf16*)alloc((size_t)Bb * Ss * Ss * 2);

    // 1) fp32 -> bf16
    cvt_f32_bf16<<<2048, 256, 0, stream>>>(x,  x_bf,  M * Dd);
    cvt_f32_bf16<<<512,  256, 0, stream>>>(Wq, wq_bf, Hh * Dd);
    cvt_f32_bf16<<<512,  256, 0, stream>>>(Wk, wk_bf, Hh * Dd);
    cvt_f32_bf16<<<512,  256, 0, stream>>>(Wv, wv_bf, Hh * Dd);

    // 2) QKV projections: (M x D) @ (H x D)^T + bias -> bf16
    dim3 gproj(Hh / 128, M / 128);
    gemm_xwT_bias<<<gproj, 256, 0, stream>>>(x_bf, wq_bf, bq, q_bf, M, Hh, Dd);
    gemm_xwT_bias<<<gproj, 256, 0, stream>>>(x_bf, wk_bf, bk, k_bf, M, Hh, Dd);
    gemm_xwT_bias<<<gproj, 256, 0, stream>>>(x_bf, wv_bf, bv, v_bf, M, Hh, Dd);

    // 3) causal softmax -> P (bf16). 2 waves/WG; LDS = 256KB scores + 64KB Q.
    const size_t smem = 2u * 16u * (size_t)Ss * 4u + 2u * 16u * (size_t)Hh * 2u;
    attn_softmax<<<M / 16 / 2, 64, smem, stream>>>(q_bf, k_bf, p_bf);

    // 4) O = P @ V (fp32 out)
    gemm_pv<<<dim3(Hh / 256, Ss / 16, Bb), 128, 0, stream>>>(p_bf, v_bf, out);
}
